// TransformerEncoderLayer_CustomAttn_49916109914147
// MI455X (gfx1250) — matrix operand order repaired and checked by reading, not compiled, not run on hardware
//
#include <hip/hip_runtime.h>

typedef __attribute__((ext_vector_type(16))) _Float16 v16h;
typedef __attribute__((ext_vector_type(8)))  _Float16 v8h;
typedef __attribute__((ext_vector_type(4)))  _Float16 v4h;
typedef __attribute__((ext_vector_type(8)))  float    v8f;
typedef __attribute__((ext_vector_type(4)))  unsigned int u32x4;
typedef __attribute__((ext_vector_type(4)))  int i32x4;
typedef __attribute__((ext_vector_type(8)))  int i32x8;

#define S_LEN 4096
#define E_DIM 256
#define P_DIM 32
#define F_DIM 1024
#define B_DIM 4

#if defined(__HIP_DEVICE_COMPILE__) && __has_builtin(__builtin_amdgcn_tensor_load_to_lds)
#define HAVE_TDM 1
#else
#define HAVE_TDM 0
#endif
#define NBUF (1 + HAVE_TDM)

__device__ __forceinline__ v8f zero8() {
  v8f z;
#pragma unroll
  for (int i = 0; i < 8; ++i) z[i] = 0.0f;
  return z;
}

__device__ __forceinline__ v8f wmma16(v16h a, v16h b, v8f c) {
  return __builtin_amdgcn_wmma_f32_16x16x32_f16(false, a, false, b, (short)0, c,
                                                false, false);
}

// A-operand fragment (16x32 f16): lane L(0-15) row=L holds K=0..7,16..23;
// lane L+16 row=L holds K=8..15,24..31.  (ISA 7.12.2)
__device__ __forceinline__ v16h load_frag_a(const _Float16* base, int ld) {
  int lane = threadIdx.x & 31;
  int row = lane & 15;
  int kb = (lane >> 4) * 8;
  const _Float16* p = base + row * ld + kb;
  v8h lo = *(const v8h*)(p);
  v8h hi = *(const v8h*)(p + 16);
  v16h a;
#pragma unroll
  for (int i = 0; i < 8; ++i) { a[i] = lo[i]; a[i + 8] = hi[i]; }
  return a;
}

// B-operand fragment (32x16 f16): lane n(0-15) col=n holds K=0..7,16..23;
// lane n+16 col=n holds K=8..15,24..31.
__device__ __forceinline__ v16h load_frag_b(const _Float16* base, int ld) {
  int lane = threadIdx.x & 31;
  int col = lane & 15;
  int kb = (lane >> 4) * 8;
  const _Float16* p = base + col * ld + kb;
  v8h lo = *(const v8h*)(p);
  v8h hi = *(const v8h*)(p + 16);
  v16h b;
#pragma unroll
  for (int i = 0; i < 8; ++i) { b[i] = lo[i]; b[i + 8] = hi[i]; }
  return b;
}

#if HAVE_TDM
// ---------------------------------------------------------------------------
// TDM 2-D tile load Global->LDS (ISA ch.8 D# descriptor, VIMAGE op 196).
// f16 elements (data_size=1 => 2B).  Optional LDS padding: +4 dwords after
// every 16 dwords => LDS row stride of 40 halves (== LDT) for 32-half rows.
// Issued by one wave; completion tracked with TENSORcnt.
// ---------------------------------------------------------------------------
__device__ __forceinline__ void tdm_load_2d_f16(
    const _Float16* gtile, void* ldsdst, unsigned td0, unsigned td1,
    unsigned long long stride0, unsigned tile0, unsigned tile1, int pad) {
  unsigned long long ga = (unsigned long long)(uintptr_t)gtile;
  unsigned lds = (unsigned)(uintptr_t)ldsdst;   // low 32b of flat = LDS offset
  u32x4 g0;
  g0[0] = 1u;                                               // count=1, user D#
  g0[1] = lds;                                              // lds_addr
  g0[2] = (unsigned)ga;                                     // global_addr lo
  g0[3] = (unsigned)((ga >> 32) & 0x01FFFFFFu) | (2u << 30);// ga hi | type=2
  unsigned w0 = 1u << 16;                                   // data_size = 2B
  if (pad) w0 |= (1u << 20) | (3u << 22) | (3u << 25);      // pad 4dw / 16dw
  i32x8 g1;
  g1[0] = (int)w0;
  g1[1] = (int)((td0 & 0xFFFFu) << 16);                     // tensor_dim0 lo
  g1[2] = (int)((td0 >> 16) | ((td1 & 0xFFFFu) << 16));     // dim0 hi|dim1 lo
  g1[3] = (int)((td1 >> 16) | (tile0 << 16));               // dim1 hi|tile0
  g1[4] = (int)tile1;                                       // tile1 | tile2=0
  g1[5] = (int)(unsigned)(stride0 & 0xFFFFFFFFu);           // dim0 stride lo
  g1[6] = (int)(unsigned)((stride0 >> 32) & 0xFFFFu);       // stride hi|s1=0
  g1[7] = 0;
  i32x4 z4;
#pragma unroll
  for (int i = 0; i < 4; ++i) z4[i] = 0;
#if defined(__clang_major__) && (__clang_major__ >= 23)
  i32x8 z8;
#pragma unroll
  for (int i = 0; i < 8; ++i) z8[i] = 0;
  __builtin_amdgcn_tensor_load_to_lds(g0, g1, z4, z4, z8, 0);
#else
  __builtin_amdgcn_tensor_load_to_lds(g0, g1, z4, z4, 0);
#endif
}
#endif  // HAVE_TDM

// ---------------------------------------------------------------------------
// pe = Wpos @ pos + bpos ; xs = pe^T / ls (f16) ; sq = ||xs||^2 per (b,s)
// ---------------------------------------------------------------------------
__global__ __launch_bounds__(256) void prep_pos_kernel(
    const float* __restrict__ pos,   // [B,P,S]
    const float* __restrict__ Wpos,  // [P,P]
    const float* __restrict__ bpos,  // [P]
    const float* __restrict__ ls,    // scalar
    _Float16* __restrict__ xs,       // [B,S,P] f16
    float* __restrict__ sq)          // [B,S]
{
  __shared__ float sW[P_DIM * P_DIM];
  __shared__ float sb[P_DIM];
  int tid = threadIdx.x;
  for (int i = tid; i < P_DIM * P_DIM; i += 256) sW[i] = Wpos[i];
  if (tid < P_DIM) sb[tid] = bpos[tid];
  __syncthreads();

  int gid = blockIdx.x * 256 + tid;   // b*S + s
  int b = gid >> 12;
  int s = gid & (S_LEN - 1);
  const float* pb = pos + (size_t)b * P_DIM * S_LEN + s;
  float x[P_DIM];
#pragma unroll
  for (int p = 0; p < P_DIM; ++p) x[p] = pb[(size_t)p * S_LEN];

  float inv_ls = 1.0f / ls[0];
  float ssum = 0.0f;
  v8h o16[4];
#pragma unroll
  for (int o = 0; o < P_DIM; ++o) {
    float acc = sb[o];
#pragma unroll
    for (int p = 0; p < P_DIM; ++p) acc = fmaf(sW[o * P_DIM + p], x[p], acc);
    acc *= inv_ls;
    ssum += acc * acc;
    o16[o >> 3][o & 7] = (_Float16)acc;
  }
  _Float16* dst = xs + (size_t)gid * P_DIM;
#pragma unroll
  for (int i = 0; i < 4; ++i) ((v8h*)dst)[i] = o16[i];
  sq[gid] = ssum;
}

// ---------------------------------------------------------------------------
// f32 -> f16 elementwise (n multiple of 1024)
// ---------------------------------------------------------------------------
__global__ __launch_bounds__(256) void f32_to_f16_kernel(
    const float* __restrict__ in, _Float16* __restrict__ out, size_t n)
{
  size_t i = ((size_t)blockIdx.x * 256 + threadIdx.x) * 4;
  if (i >= n) return;
  float4 v = *(const float4*)(in + i);
  v4h o;
  o[0] = (_Float16)v.x; o[1] = (_Float16)v.y;
  o[2] = (_Float16)v.z; o[3] = (_Float16)v.w;
  *(v4h*)(out + i) = o;
}

// ---------------------------------------------------------------------------
// Generic WMMA GEMM:  C[M,N] = A[M,K] @ W[N,K]^T (+bias)(+resid)(relu?)
// Block = 128x64 tile, 8 waves (4x2), each wave 32x32 (2x2 WMMA tiles), BK=32
// TDM double-buffered LDS staging; one barrier per k-step.
// ---------------------------------------------------------------------------
#define BM 128
#define BN 64
#define BK 32
#define LDT 40   // padded LDS row (halves) -> 80B, bank-conflict free frags

__global__ __launch_bounds__(256) void gemm_wmma_kernel(
    const _Float16* __restrict__ A, const _Float16* __restrict__ W,
    const float* __restrict__ bias, const float* __restrict__ resid,
    _Float16* __restrict__ out16, float* __restrict__ out32,
    _Float16* __restrict__ outT16,   // transposed f16 store [B][N][S] (Vt)
    int M, int N, int K, int relu)
{
  __shared__ _Float16 sA[NBUF][BM * LDT];
  __shared__ _Float16 sB[NBUF][BN * LDT];
  int tid = threadIdx.x;
  int wave = tid >> 5, lane = tid & 31;
  int bm = blockIdx.x * BM;
  int bn = blockIdx.y * BN;
  int wm = (wave >> 1) * 32;
  int wn = (wave & 1) * 32;

  v8f acc[2][2];
#pragma unroll
  for (int tm = 0; tm < 2; ++tm)
#pragma unroll
    for (int tn = 0; tn < 2; ++tn) acc[tm][tn] = zero8();

  int nk = K / BK;
  int cur = 0;
#if HAVE_TDM
  if (wave == 0) {
    tdm_load_2d_f16(A + (size_t)bm * K, &sA[0][0], (unsigned)K, BM,
                    (unsigned long long)K, BK, BM, 1);
    tdm_load_2d_f16(W + (size_t)bn * K, &sB[0][0], (unsigned)K, BN,
                    (unsigned long long)K, BK, BN, 1);
  }
#endif

  for (int kk = 0; kk < nk; ++kk) {
    int k0 = kk * BK;
#if HAVE_TDM
    if (wave == 0) __builtin_amdgcn_s_wait_tensorcnt(0);
    __syncthreads();                       // tile[cur] ready for all waves
    if (wave == 0 && kk + 1 < nk) {        // DMA next tile while we compute
      int k0n = k0 + BK;
      tdm_load_2d_f16(A + (size_t)bm * K + k0n, &sA[cur ^ 1][0], (unsigned)K,
                      BM, (unsigned long long)K, BK, BM, 1);
      tdm_load_2d_f16(W + (size_t)bn * K + k0n, &sB[cur ^ 1][0], (unsigned)K,
                      BN, (unsigned long long)K, BK, BN, 1);
    }
#else
#pragma unroll
    for (int it = 0; it < 2; ++it) {           // A tile: 128 rows x 64B
      int c = tid + it * 256;
      int row = c >> 2, seg = c & 3;
      *(v8h*)&sA[0][row * LDT + seg * 8] =
          *(const v8h*)(A + (size_t)(bm + row) * K + k0 + seg * 8);
    }
    {                                           // B tile: 64 rows x 64B
      int row = tid >> 2, seg = tid & 3;
      *(v8h*)&sB[0][row * LDT + seg * 8] =
          *(const v8h*)(W + (size_t)(bn + row) * K + k0 + seg * 8);
    }
    __syncthreads();
    if (k0 + BK < K)
      __builtin_prefetch(A + (size_t)(bm + (tid >> 1)) * K + k0 + BK, 0, 1);
#endif
#pragma unroll
    for (int tm = 0; tm < 2; ++tm) {
      v16h af = load_frag_a(&sA[cur][(wm + tm * 16) * LDT], LDT);
#pragma unroll
      for (int tn = 0; tn < 2; ++tn) {
        v16h bf = load_frag_b(&sB[cur][(wn + tn * 16) * LDT], LDT);
        acc[tm][tn] = wmma16(af, bf, acc[tm][tn]);
      }
    }
#if HAVE_TDM
    cur ^= 1;
#else
    __syncthreads();
#endif
  }

  int colL = lane & 15, rowH = (lane >> 4) * 8;
#pragma unroll
  for (int tm = 0; tm < 2; ++tm)
#pragma unroll
    for (int tn = 0; tn < 2; ++tn) {
      int n = bn + wn + tn * 16 + colL;
      float bv = bias ? bias[n] : 0.0f;
#pragma unroll
      for (int i = 0; i < 8; ++i) {
        int m = bm + wm + tm * 16 + rowH + i;
        float v = acc[tm][tn][i] + bv;
        if (relu) v = fmaxf(v, 0.0f);
        if (resid) v += resid[(size_t)m * N + n];
        size_t idx = (size_t)m * N + n;
        if (out32) out32[idx] = v;
        if (out16) out16[idx] = (_Float16)v;
        if (outT16) {
          int bb = m >> 12, ss = m & (S_LEN - 1);
          outT16[((size_t)bb * N + n) * S_LEN + ss] = (_Float16)v;
        }
      }
    }
}

// ---------------------------------------------------------------------------
// Fused RBF-kernel flash attention.
// Block: 32 query rows x all E=256 ctx cols; 8 waves each own 32 E-columns.
// Every wave computes the (redundant, K=32 -> cheap) 32x32 score tile,
// does online softmax, bounces probs C->A layout through private LDS,
// then accumulates P@V from the TDM-staged Vt LDS tile (double buffered).
// ---------------------------------------------------------------------------
__global__ __launch_bounds__(256) void flash_rbf_kernel(
    const _Float16* __restrict__ xs,   // [B,S,32] f16 (already /ls)
    const float* __restrict__ sq,      // [B,S]
    const _Float16* __restrict__ Vt,   // [B,E,S] f16
    _Float16* __restrict__ ctx16,      // [B,S,E] f16
    const float* __restrict__ os_p)    // outputscale scalar
{
  __shared__ _Float16 sV[NBUF][E_DIM * LDT];   // Vt tile: [e][32 keys] padded
  __shared__ _Float16 sP[8 * 32 * LDT];        // per-wave 32x32 prob tiles
  int tid = threadIdx.x, wave = tid >> 5, lane = tid & 31;
  int b = blockIdx.y;
  int q0 = blockIdx.x * 32;
  float oscale = os_p[0];

  const _Float16* xsb = xs + (size_t)b * S_LEN * P_DIM;
  const float* sqb = sq + (size_t)b * S_LEN;
  const _Float16* Vtb = Vt + (size_t)b * E_DIM * S_LEN;

  // Q fragments straight from global (row-contiguous, 16B chunks)
  v16h qa[2];
  {
    int row = lane & 15, kb = (lane >> 4) * 8;
#pragma unroll
    for (int tm = 0; tm < 2; ++tm) {
      const _Float16* p = xsb + (size_t)(q0 + tm * 16 + row) * P_DIM + kb;
      v8h lo = *(const v8h*)(p);
      v8h hi = *(const v8h*)(p + 16);
#pragma unroll
      for (int i = 0; i < 8; ++i) { qa[tm][i] = lo[i]; qa[tm][i + 8] = hi[i]; }
    }
  }
  float sqq[2][8];
  int rowH = (lane >> 4) * 8;
#pragma unroll
  for (int tm = 0; tm < 2; ++tm)
#pragma unroll
    for (int i = 0; i < 8; ++i) sqq[tm][i] = sqb[q0 + tm * 16 + rowH + i];

  float mrun[2][8], lrun[2][8];
#pragma unroll
  for (int tm = 0; tm < 2; ++tm)
#pragma unroll
    for (int i = 0; i < 8; ++i) { mrun[tm][i] = -1e30f; lrun[tm][i] = 0.0f; }

  v8f acc[2][2];
#pragma unroll
  for (int tm = 0; tm < 2; ++tm)
#pragma unroll
    for (int cn = 0; cn < 2; ++cn) acc[tm][cn] = zero8();

  _Float16* myP = &sP[wave * 32 * LDT];
  int cur = 0;
#if HAVE_TDM
  if (wave == 0)
    tdm_load_2d_f16(Vtb, &sV[0][0], S_LEN, E_DIM, S_LEN, 32, E_DIM, 1);
#endif

  for (int k0 = 0; k0 < S_LEN; k0 += 32) {
#if HAVE_TDM
    if (wave == 0) __builtin_amdgcn_s_wait_tensorcnt(0);
    __syncthreads();
    if (wave == 0 && k0 + 32 < S_LEN)
      tdm_load_2d_f16(Vtb + k0 + 32, &sV[cur ^ 1][0], S_LEN, E_DIM, S_LEN, 32,
                      E_DIM, 1);
#else
#pragma unroll
    for (int it = 0; it < 4; ++it) {    // [256 e][32 keys], rows of 64B
      int c = tid + it * 256;
      int e = c >> 2, seg = c & 3;
      *(v8h*)&sV[0][e * LDT + seg * 8] =
          *(const v8h*)(Vtb + (size_t)e * S_LEN + k0 + seg * 8);
    }
    __syncthreads();
#endif

    // key B-fragments straight from global (lane = key, features contiguous)
    v16h kf[2];
    {
      int col = lane & 15, kb = (lane >> 4) * 8;
#pragma unroll
      for (int tn = 0; tn < 2; ++tn) {
        const _Float16* p = xsb + (size_t)(k0 + tn * 16 + col) * P_DIM + kb;
        v8h lo = *(const v8h*)(p);
        v8h hi = *(const v8h*)(p + 16);
#pragma unroll
        for (int i = 0; i < 8; ++i) { kf[tn][i] = lo[i]; kf[tn][i + 8] = hi[i]; }
      }
    }
    // QK^T : one WMMA per 16x16 score tile (K-dim == P == 32)
    v8f sc[2][2];
#pragma unroll
    for (int tm = 0; tm < 2; ++tm)
#pragma unroll
      for (int tn = 0; tn < 2; ++tn)
        sc[tm][tn] = wmma16(qa[tm], kf[tn], zero8());

    float sqk[2];
#pragma unroll
    for (int tn = 0; tn < 2; ++tn) sqk[tn] = sqb[k0 + tn * 16 + (lane & 15)];

    // scaled RBF scores: s = os * exp(-0.5 * max(d2, 0))
#pragma unroll
    for (int tm = 0; tm < 2; ++tm)
#pragma unroll
      for (int tn = 0; tn < 2; ++tn)
#pragma unroll
        for (int i = 0; i < 8; ++i) {
          float d2 = sqq[tm][i] + sqk[tn] - 2.0f * sc[tm][tn][i];
          d2 = fmaxf(d2, 0.0f);
          sc[tm][tn][i] = oscale * __expf(-0.5f * d2);
        }

    // online softmax; xor masks 1..8 reduce within each 16-lane half (== row)
#pragma unroll
    for (int tm = 0; tm < 2; ++tm)
#pragma unroll
      for (int i = 0; i < 8; ++i) {
        float rm = fmaxf(sc[tm][0][i], sc[tm][1][i]);
#pragma unroll
        for (int m = 1; m <= 8; m <<= 1) rm = fmaxf(rm, __shfl_xor(rm, m, 32));
        float mnew = fmaxf(mrun[tm][i], rm);
        float scale = __expf(mrun[tm][i] - mnew);
        float p0 = __expf(sc[tm][0][i] - mnew);
        float p1 = __expf(sc[tm][1][i] - mnew);
        float rs = p0 + p1;
#pragma unroll
        for (int m = 1; m <= 8; m <<= 1) rs += __shfl_xor(rs, m, 32);
        lrun[tm][i] = lrun[tm][i] * scale + rs;
        mrun[tm][i] = mnew;
#pragma unroll
        for (int cn = 0; cn < 2; ++cn) acc[tm][cn][i] *= scale;
        int prow = tm * 16 + rowH + i;
        myP[prow * LDT + (lane & 15)] = (_Float16)p0;
        myP[prow * LDT + 16 + (lane & 15)] = (_Float16)p1;
      }

    // probs C-layout -> A-layout via wave-private LDS, then P@V
    v16h pa[2];
#pragma unroll
    for (int tm = 0; tm < 2; ++tm)
      pa[tm] = load_frag_a(&myP[tm * 16 * LDT], LDT);
#pragma unroll
    for (int cn = 0; cn < 2; ++cn) {
      v16h vb = load_frag_b(&sV[cur][(wave * 32 + cn * 16) * LDT], LDT);
#pragma unroll
      for (int tm = 0; tm < 2; ++tm)
        acc[tm][cn] = wmma16(pa[tm], vb, acc[tm][cn]);
    }
#if HAVE_TDM
    cur ^= 1;
#else
    __syncthreads();
#endif
  }

  // finalize: ctx = acc / l
#pragma unroll
  for (int tm = 0; tm < 2; ++tm)
#pragma unroll
    for (int cn = 0; cn < 2; ++cn)
#pragma unroll
      for (int i = 0; i < 8; ++i) {
        int m = q0 + tm * 16 + rowH + i;
        int e = wave * 32 + cn * 16 + (lane & 15);
        ctx16[((size_t)b * S_LEN + m) * E_DIM + e] =
            (_Float16)(acc[tm][cn][i] / lrun[tm][i]);
      }
}

// ---------------------------------------------------------------------------
// LayerNorm over E=256; one wave per row (lane holds 8 elems), 8 rows/block.
// ---------------------------------------------------------------------------
__global__ __launch_bounds__(256) void layernorm_kernel(
    const float* __restrict__ x, const float* __restrict__ g,
    const float* __restrict__ bt, float* __restrict__ out32,
    _Float16* __restrict__ out16)
{
  int wave = threadIdx.x >> 5, lane = threadIdx.x & 31;
  int row = blockIdx.x * 8 + wave;
  const float* xr = x + (size_t)row * E_DIM;
  float v[8];
  *(float4*)&v[0] = *(const float4*)(xr + lane * 8);
  *(float4*)&v[4] = *(const float4*)(xr + lane * 8 + 4);
  float s = 0.0f;
#pragma unroll
  for (int i = 0; i < 8; ++i) s += v[i];
#pragma unroll
  for (int m = 1; m <= 16; m <<= 1) s += __shfl_xor(s, m, 32);
  float mu = s * (1.0f / E_DIM);
  float s2 = 0.0f;
#pragma unroll
  for (int i = 0; i < 8; ++i) { float d = v[i] - mu; s2 += d * d; }
#pragma unroll
  for (int m = 1; m <= 16; m <<= 1) s2 += __shfl_xor(s2, m, 32);
  float rstd = rsqrtf(s2 * (1.0f / E_DIM) + 1e-5f);
#pragma unroll
  for (int i = 0; i < 8; ++i) {
    int e = lane * 8 + i;
    float y = (v[i] - mu) * rstd * g[e] + bt[e];
    size_t idx = (size_t)row * E_DIM + e;
    if (out32) out32[idx] = y;
    if (out16) out16[idx] = (_Float16)y;
  }
}

// ---------------------------------------------------------------------------
extern "C" void kernel_launch(void* const* d_in, const int* in_sizes, int n_in,
                              void* d_out, int out_size, void* d_ws,
                              size_t ws_size, hipStream_t stream) {
  const float* src  = (const float*)d_in[0];
  const float* pos  = (const float*)d_in[1];
  const float* Wpos = (const float*)d_in[2];
  const float* bpos = (const float*)d_in[3];
  const float* ls   = (const float*)d_in[4];
  const float* os   = (const float*)d_in[5];
  const float* Wv   = (const float*)d_in[6];
  const float* bv   = (const float*)d_in[7];
  const float* Wo   = (const float*)d_in[8];
  const float* bo   = (const float*)d_in[9];
  const float* W1   = (const float*)d_in[10];
  const float* b1   = (const float*)d_in[11];
  const float* W2   = (const float*)d_in[12];
  const float* b2   = (const float*)d_in[13];
  const float* g1   = (const float*)d_in[14];
  const float* be1  = (const float*)d_in[15];
  const float* g2   = (const float*)d_in[16];
  const float* be2  = (const float*)d_in[17];

  const size_t M = (size_t)B_DIM * S_LEN;   // 16384
  char* w = (char*)d_ws;
  size_t off = 0;
  auto alloc = [&](size_t bytes) { void* p = w + off; off += (bytes + 255) & ~(size_t)255; return p; };

  _Float16* xs16  = (_Float16*)alloc(M * P_DIM * 2);
  float*    sqv   = (float*)   alloc(M * 4);
  _Float16* src16 = (_Float16*)alloc(M * E_DIM * 2);
  _Float16* Wv16  = (_Float16*)alloc((size_t)E_DIM * E_DIM * 2);
  _Float16* Wo16  = (_Float16*)alloc((size_t)E_DIM * E_DIM * 2);
  _Float16* W116  = (_Float16*)alloc((size_t)F_DIM * E_DIM * 2);
  _Float16* W216  = (_Float16*)alloc((size_t)E_DIM * F_DIM * 2);
  _Float16* Vt16  = (_Float16*)alloc((size_t)B_DIM * E_DIM * S_LEN * 2);
  _Float16* ctx16 = (_Float16*)alloc(M * E_DIM * 2);
  float*    xpre  = (float*)   alloc(M * E_DIM * 4);   // also reused as ypre
  float*    x32   = (float*)   alloc(M * E_DIM * 4);
  _Float16* x16   = (_Float16*)alloc(M * E_DIM * 2);
  _Float16* h16   = (_Float16*)alloc(M * F_DIM * 2);
  float*    ypre  = xpre;                               // alias (xpre dead)
  (void)ws_size; (void)n_in; (void)in_sizes; (void)out_size;

  // 1. positional transform -> xs (f16) + sq
  prep_pos_kernel<<<dim3(M / 256), dim3(256), 0, stream>>>(pos, Wpos, bpos, ls,
                                                           xs16, sqv);
  // 2. f32 -> f16 conversions
  f32_to_f16_kernel<<<dim3((M * E_DIM) / 1024), 256, 0, stream>>>(src, src16, M * E_DIM);
  f32_to_f16_kernel<<<dim3((E_DIM * E_DIM) / 1024), 256, 0, stream>>>(Wv, Wv16, (size_t)E_DIM * E_DIM);
  f32_to_f16_kernel<<<dim3((E_DIM * E_DIM) / 1024), 256, 0, stream>>>(Wo, Wo16, (size_t)E_DIM * E_DIM);
  f32_to_f16_kernel<<<dim3((F_DIM * E_DIM) / 1024), 256, 0, stream>>>(W1, W116, (size_t)F_DIM * E_DIM);
  f32_to_f16_kernel<<<dim3((E_DIM * F_DIM) / 1024), 256, 0, stream>>>(W2, W216, (size_t)E_DIM * F_DIM);

  // 3. V = src @ Wv^T + bv, stored transposed -> Vt [B,E,S]
  gemm_wmma_kernel<<<dim3(M / BM, E_DIM / BN), 256, 0, stream>>>(
      src16, Wv16, bv, nullptr, nullptr, nullptr, Vt16, (int)M, E_DIM, E_DIM, 0);

  // 4. fused RBF-score / softmax / P@V  -> ctx (f16)
  flash_rbf_kernel<<<dim3(S_LEN / 32, B_DIM), 256, 0, stream>>>(xs16, sqv, Vt16,
                                                                ctx16, os);
  // 5. attn_out = ctx @ Wo^T + bo, + src residual -> xpre (f32)
  gemm_wmma_kernel<<<dim3(M / BM, E_DIM / BN), 256, 0, stream>>>(
      ctx16, Wo16, bo, src, nullptr, xpre, nullptr, (int)M, E_DIM, E_DIM, 0);

  // 6. LN1 -> x32 / x16
  layernorm_kernel<<<dim3(M / 8), 256, 0, stream>>>(xpre, g1, be1, x32, x16);

  // 7. h = relu(x @ W1^T + b1) -> h16
  gemm_wmma_kernel<<<dim3(M / BM, F_DIM / BN), 256, 0, stream>>>(
      x16, W116, b1, nullptr, h16, nullptr, nullptr, (int)M, F_DIM, E_DIM, 1);

  // 8. y = h @ W2^T + b2, + x residual -> ypre (f32)
  gemm_wmma_kernel<<<dim3(M / BM, E_DIM / BN), 256, 0, stream>>>(
      h16, W216, b2, x32, nullptr, ypre, nullptr, (int)M, E_DIM, F_DIM, 0);

  // 9. LN2 -> d_out (f32)
  layernorm_kernel<<<dim3(M / 8), 256, 0, stream>>>(ypre, g2, be2, (float*)d_out,
                                                    nullptr);
}
